// GroupedQueryAttention_14147622273067
// MI455X (gfx1250) — compile-verified, matrix operand order checked by reading, not run on hardware
//
#include <hip/hip_runtime.h>
#include <hip/hip_bf16.h>
#include <stdint.h>

// ---------------------------------------------------------------- constants
#define D_MODEL 2048
#define T_SEQ   2048
#define BATCH   2
#define NHEADS  32
#define NKV     8
#define DK      64
#define WINDOW  512
#define MROWS   (BATCH * T_SEQ)   // 4096
#define KVD     (NKV * DK)        // 512

typedef __attribute__((ext_vector_type(16))) __bf16 v16bf;
typedef __attribute__((ext_vector_type(8)))  float  v8f;

union FragBF { v16bf v; uint4 q[2]; __bf16 e[16]; };

// CDNA5 async global->LDS copy path (ASYNCcnt-tracked), with safe fallback.
// The builtin's parameter type (from clang's diagnostic) is a pointer to a
// GCC-style int 4-vector: global source in AS1, LDS dest in AS3.
#if defined(__HIP_DEVICE_COMPILE__) && __has_builtin(__builtin_amdgcn_global_load_async_to_lds_b128)
#define HAVE_ASYNC_LDS 1
typedef int v4i_b128 __attribute__((vector_size(16)));
typedef __attribute__((address_space(1))) v4i_b128 as1_b128;
typedef __attribute__((address_space(3))) v4i_b128 as3_b128;
#if __has_builtin(__builtin_amdgcn_s_wait_asynccnt)
#define WAIT_ASYNC() __builtin_amdgcn_s_wait_asynccnt(0)
#else
#define WAIT_ASYNC() asm volatile("s_wait_asynccnt 0x0" ::: "memory")
#endif
#endif

__device__ inline __bf16 f2bf(float f) {
    union { float f; uint32_t u; } v; v.f = f;
    uint32_t r = v.u + 0x7FFFu + ((v.u >> 16) & 1u);   // round-to-nearest-even
    union { uint16_t u; __bf16 b; } o; o.u = (uint16_t)(r >> 16);
    return o.b;
}

__device__ inline v8f wmma_bf16(v16bf a, v16bf b, v8f c) {
    // emits v_wmma_f32_16x16x32_bf16
    return __builtin_amdgcn_wmma_f32_16x16x32_bf16(false, a, false, b,
                                                   (short)0, c, false, false);
}

// ---------------------------------------------------------------- fp32 -> bf16
__global__ void cvt_f32_bf16(const float* __restrict__ in,
                             __bf16* __restrict__ out, int n4) {
    int i = blockIdx.x * blockDim.x + threadIdx.x;
    if (i < n4) {
        float4 f = ((const float4*)in)[i];
        union { __bf16 b[4]; uint64_t u; } o;
        o.b[0] = f2bf(f.x); o.b[1] = f2bf(f.y);
        o.b[2] = f2bf(f.z); o.b[3] = f2bf(f.w);
        ((uint64_t*)out)[i] = o.u;
    }
}

// ---------------------------------------------------------------- GEMM
// C[M,N] = A[M,K] * W[N,K]^T   (both K-contiguous row-major)
// MODE: 0 = bf16 out, 1 = bf16 out + fused RoPE, 2 = f32 out
// Block: 256 thr = 8 waves in 4(M) x 2(N); wave tile 32x64; block tile 128x128.
template<int MODE>
__global__ __launch_bounds__(256) void gemm_bf16_wmma(
    const __bf16* __restrict__ A, const __bf16* __restrict__ W,
    void* __restrict__ Cout, int M, int N, int K)
{
    const int lane  = threadIdx.x & 31;
    const int w     = threadIdx.x >> 5;
    const int wm    = w & 3, wn = w >> 2;
    const int m0    = blockIdx.y * 128 + wm * 32;
    const int n0    = blockIdx.x * 128 + wn * 64;
    const int lr    = lane & 15;
    const int kc    = (lane >> 4) * 8;        // 8-elem K chunk base per half-wave
    const int half8 = (lane >> 4) << 3;

    const v8f vzero = {0.f,0.f,0.f,0.f,0.f,0.f,0.f,0.f};
    v8f acc[2][4];
    #pragma unroll
    for (int i = 0; i < 2; i++)
        #pragma unroll
        for (int j = 0; j < 4; j++) acc[i][j] = vzero;

    const __bf16* Arow0 = A + (size_t)(m0 + lr)      * K + kc;
    const __bf16* Arow1 = A + (size_t)(m0 + 16 + lr) * K + kc;
    const __bf16* Wrow[4];
    #pragma unroll
    for (int nt = 0; nt < 4; nt++)
        Wrow[nt] = W + (size_t)(n0 + nt * 16 + lr) * K + kc;

    for (int k0 = 0; k0 < K; k0 += 32) {
        FragBF a0, a1;
        a0.q[0] = *(const uint4*)(Arow0 + k0);
        a0.q[1] = *(const uint4*)(Arow0 + k0 + 16);
        a1.q[0] = *(const uint4*)(Arow1 + k0);
        a1.q[1] = *(const uint4*)(Arow1 + k0 + 16);
        __builtin_prefetch(Arow0 + k0 + 128, 0, 1);   // global_prefetch_b8
        #pragma unroll
        for (int nt = 0; nt < 4; nt++) {
            FragBF b;
            b.q[0] = *(const uint4*)(Wrow[nt] + k0);
            b.q[1] = *(const uint4*)(Wrow[nt] + k0 + 16);
            acc[0][nt] = wmma_bf16(a0.v, b.v, acc[0][nt]);
            acc[1][nt] = wmma_bf16(a1.v, b.v, acc[1][nt]);
        }
    }

    if constexpr (MODE == 1) {
        // Fused RoPE: wave's 64 N-cols = exactly one head; col c pairs with
        // c+32 => fragment nt pairs with nt+2, same lane/row. Pure elementwise.
        #pragma unroll
        for (int mt = 0; mt < 2; mt++) {
            #pragma unroll
            for (int pr = 0; pr < 2; pr++) {
                int   gn   = n0 + pr * 16 + lr;
                int   j    = gn & 31;                       // freq index
                float invf = __expf(-0.28782313662425572f * (float)j); // 10000^(-j/32)
                #pragma unroll
                for (int r = 0; r < 8; r++) {
                    int   gm = m0 + mt * 16 + r + half8;
                    float t  = (float)(gm & (T_SEQ - 1));   // position in sequence
                    float s, c;
                    __sincosf(t * invf, &s, &c);
                    float f0 = acc[mt][pr][r], f2 = acc[mt][pr + 2][r];
                    acc[mt][pr][r]     = f0 * c - f2 * s;
                    acc[mt][pr + 2][r] = f2 * c + f0 * s;
                }
            }
        }
    }

    #pragma unroll
    for (int mt = 0; mt < 2; mt++)
        #pragma unroll
        for (int nt = 0; nt < 4; nt++)
            #pragma unroll
            for (int r = 0; r < 8; r++) {
                int gm = m0 + mt * 16 + r + half8;
                int gn = n0 + nt * 16 + lr;
                if constexpr (MODE == 2)
                    ((float*)Cout)[(size_t)gm * N + gn] = acc[mt][nt][r];
                else
                    ((__bf16*)Cout)[(size_t)gm * N + gn] = f2bf(acc[mt][nt][r]);
            }
}

// ---------------------------------------------------------------- attention
// One workgroup = (batch b, query head hq, 64-query block). 4 waves x 16 q.
// Flash-style online softmax over the sliding window, 32 keys per tile.
// K tile is staged once per block into LDS via the CDNA5 async global->LDS
// copy (ASYNCcnt); V tile is staged transposed for contiguous PV B-fragments.
__global__ __launch_bounds__(128) void attn_swa_gqa(
    const __bf16* __restrict__ Qb, const __bf16* __restrict__ Kb,
    const __bf16* __restrict__ Vb, __bf16* __restrict__ Ob)
{
    __shared__ __attribute__((aligned(16))) __bf16 Kt[32 * 64];    // [key][dk] row-major
    __shared__ __attribute__((aligned(16))) __bf16 VtT[DK * 40];   // [dk][key], stride 40
    __shared__ __attribute__((aligned(16))) __bf16 Pt[4][16 * 32]; // per-wave P tile

    const int lane  = threadIdx.x & 31;
    const int w     = threadIdx.x >> 5;
    const int lr    = lane & 15;
    const int kc    = (lane >> 4) * 8;
    const int half8 = (lane >> 4) << 3;

    const int    qblk = blockIdx.x;
    const int    hq   = blockIdx.y;
    const int    b    = blockIdx.z;
    const int    kvh  = hq >> 2;                 // GQA: 4 q-heads per kv-head
    const int    q0b  = qblk * 64;
    const int    q0   = q0b + w * 16;
    const size_t bT   = (size_t)b * T_SEQ;

    // Q fragments (16 queries x 64 dk), loaded once
    FragBF aq[2];
    {
        const __bf16* qp = Qb + (bT + q0 + lr) * D_MODEL + (size_t)hq * DK + kc;
        aq[0].q[0] = *(const uint4*)(qp);
        aq[0].q[1] = *(const uint4*)(qp + 16);
        aq[1].q[0] = *(const uint4*)(qp + 32);
        aq[1].q[1] = *(const uint4*)(qp + 48);
    }

    const v8f vzero = {0.f,0.f,0.f,0.f,0.f,0.f,0.f,0.f};
    v8f o[4];
    #pragma unroll
    for (int nt = 0; nt < 4; nt++) o[nt] = vzero;
    float mrow[8], ssum[8];
    #pragma unroll
    for (int r = 0; r < 8; r++) { mrow[r] = -1e30f; ssum[r] = 0.0f; }

    int kbeg = q0b - (WINDOW - 1); if (kbeg < 0) kbeg = 0; kbeg &= ~31;
    const int kend = q0b + 64;                   // exclusive; tiles cover exactly

    for (int k0 = kbeg; k0 < kend; k0 += 32) {
        __syncthreads();
        // ---- stage K tile (32 keys x 64 dk) row-major into LDS
        {
            int row = threadIdx.x >> 2;          // key row 0..31
            int ck  = (threadIdx.x & 3) * 16;    // 16 dk elems (32B) per thread
            const __bf16* kp = Kb + (bT + k0 + row) * KVD + (size_t)kvh * DK + ck;
            __bf16* lp = &Kt[row * 64 + ck];
#ifdef HAVE_ASYNC_LDS
            __builtin_amdgcn_global_load_async_to_lds_b128(
                (as1_b128*)(uintptr_t)kp,
                (as3_b128*)(uint32_t)(uintptr_t)lp, 0, 0);
            __builtin_amdgcn_global_load_async_to_lds_b128(
                (as1_b128*)(uintptr_t)(kp + 8),
                (as3_b128*)(uint32_t)(uintptr_t)(lp + 8), 0, 0);
#else
            uint4 c0 = *(const uint4*)kp;
            uint4 c1 = *(const uint4*)(kp + 8);
            *(uint4*)lp = c0;
            *(uint4*)(lp + 8) = c1;
#endif
        }
        // ---- stage V tile TRANSPOSED into LDS: VtT[dk][key]
        {
            int row = threadIdx.x >> 2;          // key row 0..31
            int dkc = (threadIdx.x & 3) * 16;    // 16 dk elems per thread
            FragBF vv;
            const __bf16* vp = Vb + (bT + k0 + row) * KVD + (size_t)kvh * DK + dkc;
            vv.q[0] = *(const uint4*)(vp);
            vv.q[1] = *(const uint4*)(vp + 8);
            #pragma unroll
            for (int i = 0; i < 16; i++) VtT[(dkc + i) * 40 + row] = vv.e[i];
        }
#ifdef HAVE_ASYNC_LDS
        WAIT_ASYNC();                            // s_wait_asynccnt 0
#endif
        __syncthreads();

        // ---- S = Q * K^T  (16 x 32 scores, f32 accum), K from LDS
        v8f s0 = vzero, s1 = vzero;
        {
            const __bf16* kr0 = &Kt[lr * 64 + kc];          // keys 0..15
            const __bf16* kr1 = &Kt[(16 + lr) * 64 + kc];   // keys 16..31
            FragBF bk;
            bk.q[0] = *(const uint4*)(kr0);       bk.q[1] = *(const uint4*)(kr0 + 16);
            s0 = wmma_bf16(aq[0].v, bk.v, s0);
            bk.q[0] = *(const uint4*)(kr0 + 32);  bk.q[1] = *(const uint4*)(kr0 + 48);
            s0 = wmma_bf16(aq[1].v, bk.v, s0);
            bk.q[0] = *(const uint4*)(kr1);       bk.q[1] = *(const uint4*)(kr1 + 16);
            s1 = wmma_bf16(aq[0].v, bk.v, s1);
            bk.q[0] = *(const uint4*)(kr1 + 32);  bk.q[1] = *(const uint4*)(kr1 + 48);
            s1 = wmma_bf16(aq[1].v, bk.v, s1);
        }

        // ---- masked online softmax (row = query, 16 lanes hold 16 keys)
        constexpr float scl = 0.125f;            // 1/sqrt(64)
        #pragma unroll
        for (int r = 0; r < 8; r++) {
            int  q    = q0 + r + half8;
            int  key0 = k0 + lr;
            int  d0   = q - key0, d1 = q - key0 - 16;
            bool vv0  = (d0 >= 0) && (d0 < WINDOW);
            bool vv1  = (d1 >= 0) && (d1 < WINDOW);
            float a0  = vv0 ? s0[r] * scl : -1e30f;
            float a1  = vv1 ? s1[r] * scl : -1e30f;
            float mt  = fmaxf(a0, a1);
            #pragma unroll
            for (int x = 1; x < 16; x <<= 1) mt = fmaxf(mt, __shfl_xor(mt, x, 32));
            float mn    = fmaxf(mrow[r], mt);
            float alpha = __expf(mrow[r] - mn);
            float p0    = vv0 ? __expf(a0 - mn) : 0.0f;
            float p1    = vv1 ? __expf(a1 - mn) : 0.0f;
            float rs    = p0 + p1;
            #pragma unroll
            for (int x = 1; x < 16; x <<= 1) rs += __shfl_xor(rs, x, 32);
            ssum[r] = ssum[r] * alpha + rs;
            mrow[r] = mn;
            o[0][r] *= alpha; o[1][r] *= alpha; o[2][r] *= alpha; o[3][r] *= alpha;
            // P (C-layout) -> LDS so it can be re-read in A-fragment layout
            Pt[w][(r + half8) * 32 + lr]      = f2bf(p0);
            Pt[w][(r + half8) * 32 + 16 + lr] = f2bf(p1);
        }

        // ---- O += P * V   (contraction over 32 keys, N = 64 dk)
        FragBF pf;
        pf.q[0] = *(const uint4*)&Pt[w][lr * 32 + kc];
        pf.q[1] = *(const uint4*)&Pt[w][lr * 32 + kc + 16];
        #pragma unroll
        for (int nt = 0; nt < 4; nt++) {
            FragBF vf;
            const __bf16* vb = &VtT[(nt * 16 + lr) * 40 + kc];
            vf.q[0] = *(const uint4*)(vb);
            vf.q[1] = *(const uint4*)(vb + 16);
            o[nt] = wmma_bf16(pf.v, vf.v, o[nt]);
        }
    }

    // ---- normalize and write attention output (bf16)
    #pragma unroll
    for (int nt = 0; nt < 4; nt++)
        #pragma unroll
        for (int r = 0; r < 8; r++) {
            int   q   = q0 + r + half8;
            float val = o[nt][r] / ssum[r];
            Ob[(bT + q) * D_MODEL + (size_t)hq * DK + nt * 16 + lr] = f2bf(val);
        }
}

// ---------------------------------------------------------------- launch
extern "C" void kernel_launch(void* const* d_in, const int* in_sizes, int n_in,
                              void* d_out, int out_size, void* d_ws, size_t ws_size,
                              hipStream_t stream)
{
    const float* x  = (const float*)d_in[0];
    const float* Wq = (const float*)d_in[1];
    const float* Wk = (const float*)d_in[2];
    const float* Wv = (const float*)d_in[3];
    const float* Wo = (const float*)d_in[4];

    char* ws = (char*)d_ws;
    __bf16* xb  = (__bf16*)(ws + 0);           // 4096x2048    16.0 MiB
    __bf16* Wqb = (__bf16*)(ws + 16777216);    // 2048x2048     8.0 MiB
    __bf16* Wkb = (__bf16*)(ws + 25165824);    //  512x2048     2.0 MiB
    __bf16* Wvb = (__bf16*)(ws + 27262976);    //  512x2048     2.0 MiB
    __bf16* Wob = (__bf16*)(ws + 29360128);    // 2048x2048     8.0 MiB
    __bf16* Qb  = (__bf16*)(ws + 37748736);    // 4096x2048    16.0 MiB
    __bf16* Kbf = (__bf16*)(ws + 54525952);    // 4096x512      4.0 MiB
    __bf16* Vbf = (__bf16*)(ws + 58720256);    // 4096x512      4.0 MiB
    __bf16* Ob  = (__bf16*)(ws + 62914560);    // 4096x2048    16.0 MiB

    // fp32 -> bf16 conversions
    {
        int n4;
        n4 = (MROWS * D_MODEL) / 4;
        cvt_f32_bf16<<<(n4 + 255) / 256, 256, 0, stream>>>(x, xb, n4);
        n4 = (D_MODEL * D_MODEL) / 4;
        cvt_f32_bf16<<<(n4 + 255) / 256, 256, 0, stream>>>(Wq, Wqb, n4);
        cvt_f32_bf16<<<(n4 + 255) / 256, 256, 0, stream>>>(Wo, Wob, n4);
        n4 = (KVD * D_MODEL) / 4;
        cvt_f32_bf16<<<(n4 + 255) / 256, 256, 0, stream>>>(Wk, Wkb, n4);
        cvt_f32_bf16<<<(n4 + 255) / 256, 256, 0, stream>>>(Wv, Wvb, n4);
    }

    // Projections (RoPE fused into Q and K epilogues)
    gemm_bf16_wmma<1><<<dim3(D_MODEL / 128, MROWS / 128), 256, 0, stream>>>(
        xb, Wqb, Qb, MROWS, D_MODEL, D_MODEL);
    gemm_bf16_wmma<1><<<dim3(KVD / 128, MROWS / 128), 256, 0, stream>>>(
        xb, Wkb, Kbf, MROWS, KVD, D_MODEL);
    gemm_bf16_wmma<0><<<dim3(KVD / 128, MROWS / 128), 256, 0, stream>>>(
        xb, Wvb, Vbf, MROWS, KVD, D_MODEL);

    // Sliding-window GQA attention
    attn_swa_gqa<<<dim3(T_SEQ / 64, NHEADS, BATCH), 128, 0, stream>>>(
        Qb, Kbf, Vbf, Ob);

    // Output projection, f32 result
    gemm_bf16_wmma<2><<<dim3(D_MODEL / 128, MROWS / 128), 256, 0, stream>>>(
        Ob, Wob, d_out, MROWS, D_MODEL, D_MODEL);
}
